// VisionMamba_66503273611836
// MI455X (gfx1250) — compile-verified, matrix operand order checked by reading, not run on hardware
//
#include <hip/hip_runtime.h>
#include <hip/hip_bf16.h>

typedef __attribute__((ext_vector_type(16))) _Float16 v16h;
typedef __attribute__((ext_vector_type(8)))  _Float16 v8h;
typedef __attribute__((ext_vector_type(8)))  float    v8f;

#define BB   4
#define IMG  224
#define PP   16
#define DD   384
#define DEPTH 20
#define DI   768
#define SS   16
#define DTR  24
#define KK   4
#define LL   196      // (224/16)^2
#define NCLS 1000
#define BL   (BB*LL)  // 784 = 49 * 16, tile-exact in M
#define XDP  64       // x_proj output padded 56 -> 64
#define DTKP 32       // dt_proj K padded 24 -> 32

// ---------------------------------------------------------------------------
// Patch embed: hidden[b,l,d] = conv16x16/s16(x)[b,d,l] + patch_b[d] + pos[l,d]
// ---------------------------------------------------------------------------
__global__ void vm_patch_embed(const float* __restrict__ x,
                               const float* __restrict__ pw,
                               const float* __restrict__ pb,
                               const float* __restrict__ pos,
                               float* __restrict__ hidden) {
    int idx = blockIdx.x * blockDim.x + threadIdx.x;
    if (idx >= BL * DD) return;
    int d = idx % DD;
    int l = (idx / DD) % LL;
    int b = idx / (DD * LL);
    int ph = l / 14, pcol = l % 14;
    float s = pb[d];
    for (int c = 0; c < 3; ++c)
        for (int i = 0; i < PP; ++i) {
            const float* xr = x + (((size_t)(b * 3 + c) * IMG) + (ph * PP + i)) * IMG + pcol * PP;
            const float* wr = pw + (((size_t)(d * 3 + c) * PP) + i) * PP;
#pragma unroll
            for (int j = 0; j < PP; ++j) s += xr[j] * wr[j];
        }
    hidden[idx] = s + pos[(size_t)l * DD + d];
}

__global__ void vm_zero(float* __restrict__ p, int n) {
    int i = blockIdx.x * blockDim.x + threadIdx.x;
    if (i < n) p[i] = 0.f;
}

// ---------------------------------------------------------------------------
// Weight prep: fp32 -> fp16 (and zero-padding for the two ragged shapes).
// ---------------------------------------------------------------------------
__global__ void vm_cvt_f16(const float* __restrict__ s, _Float16* __restrict__ d, int n) {
    int i = blockIdx.x * blockDim.x + threadIdx.x;
    if (i < n) d[i] = (_Float16)s[i];
}
__global__ void vm_pad_xproj(const float* __restrict__ s, _Float16* __restrict__ d) {
    int idx = blockIdx.x * blockDim.x + threadIdx.x;     // [56,768] -> [64,768]
    if (idx >= XDP * DI) return;
    int k = idx % DI, n = idx / DI;
    d[idx] = (n < DTR + 2 * SS) ? (_Float16)s[(size_t)n * DI + k] : (_Float16)0.f;
}
__global__ void vm_pad_dtw(const float* __restrict__ s, _Float16* __restrict__ d) {
    int idx = blockIdx.x * blockDim.x + threadIdx.x;     // [768,24] -> [768,32]
    if (idx >= DI * DTKP) return;
    int k = idx % DTKP, n = idx / DTKP;
    d[idx] = (k < DTR) ? (_Float16)s[(size_t)n * DTR + k] : (_Float16)0.f;
}

// ---------------------------------------------------------------------------
// Fused residual-add + LayerNorm over D=384; writes f32 + f16 normed output.
// ---------------------------------------------------------------------------
__global__ __launch_bounds__(128) void vm_add_ln(float* __restrict__ res,
                                                 const float* __restrict__ hid,
                                                 float* __restrict__ xn,
                                                 _Float16* __restrict__ xn16,
                                                 const float* __restrict__ w,
                                                 const float* __restrict__ bsh,
                                                 int store_res) {
    __shared__ float sbuf[128];
    int row = blockIdx.x, t = threadIdx.x;
    size_t base = (size_t)row * DD;
    float v[3], s = 0.f;
#pragma unroll
    for (int i = 0; i < 3; ++i) {
        int c = t + i * 128;
        v[i] = res[base + c] + hid[base + c];
        if (store_res) res[base + c] = v[i];
        s += v[i];
    }
    sbuf[t] = s; __syncthreads();
    for (int o = 64; o > 0; o >>= 1) { if (t < o) sbuf[t] += sbuf[t + o]; __syncthreads(); }
    float mean = sbuf[0] * (1.f / DD);
    __syncthreads();
    float s2 = 0.f;
#pragma unroll
    for (int i = 0; i < 3; ++i) { float dd = v[i] - mean; s2 += dd * dd; }
    sbuf[t] = s2; __syncthreads();
    for (int o = 64; o > 0; o >>= 1) { if (t < o) sbuf[t] += sbuf[t + o]; __syncthreads(); }
    float inv = rsqrtf(sbuf[0] * (1.f / DD) + 1e-5f);
#pragma unroll
    for (int i = 0; i < 3; ++i) {
        int c = t + i * 128;
        float o = (v[i] - mean) * inv * w[c] + bsh[c];
        xn[base + c] = o;
        xn16[base + c] = (_Float16)o;
    }
}

// ---------------------------------------------------------------------------
// Register-blocked WMMA GEMM: C[784,N] = A[784,K] x W[N,K]^T.
// Each wave computes a 16x64 output strip (4 n-tiles): per k-step it loads
// 1 A fragment + 4 B fragments (10 x b128) and issues 4 independent WMMAs.
// All N used are multiples of 64 -> no in-wave guards; whole-wave early exit
// handles grid over-coverage. f16 operands, f32 accumulate.
// mode&1: softplus(acc + bias[n]);  mode&2: also write f16 copy of C.
// ---------------------------------------------------------------------------
__global__ __launch_bounds__(128) void vm_wmma_gemm(const _Float16* __restrict__ A,
                                                    const _Float16* __restrict__ W,
                                                    const float* __restrict__ bias,
                                                    float* __restrict__ C,
                                                    _Float16* __restrict__ C16,
                                                    int N, int K, int lda, int mode) {
    const int lane  = threadIdx.x & 31;
    const int wave  = threadIdx.x >> 5;
    const int mBase = blockIdx.x * 16;
    const int nBase = (blockIdx.y * 4 + wave) * 64;   // 4 n-tiles per wave
    if (nBase >= N) return;                           // uniform per wave
    const int half_ = lane >> 4;
    const int mr    = lane & 15;
    const _Float16* Ap = A + (size_t)(mBase + mr) * lda + half_ * 8;
    const _Float16* Wp = W + (size_t)(nBase + mr) * K + half_ * 8;
    const size_t ws16 = (size_t)16 * K;               // stride between n-tiles

    v8f acc[4] = {v8f{}, v8f{}, v8f{}, v8f{}};
    for (int kb = 0; kb < K; kb += 32) {
        __builtin_prefetch(Ap + kb + 32, 0, 1);       // global_prefetch_b8
        __builtin_prefetch(Wp + kb + 32, 0, 1);
        v8h a_lo = *(const v8h*)(Ap + kb);
        v8h a_hi = *(const v8h*)(Ap + kb + 16);
        v16h af = __builtin_shufflevector(a_lo, a_hi,
                      0,1,2,3,4,5,6,7,8,9,10,11,12,13,14,15);
#pragma unroll
        for (int j = 0; j < 4; ++j) {
            const _Float16* Wj = Wp + (size_t)j * ws16 + kb;
            v8h b_lo = *(const v8h*)(Wj);
            v8h b_hi = *(const v8h*)(Wj + 16);
            v16h bf = __builtin_shufflevector(b_lo, b_hi,
                          0,1,2,3,4,5,6,7,8,9,10,11,12,13,14,15);
            acc[j] = __builtin_amdgcn_wmma_f32_16x16x32_f16(false, af, false, bf,
                                                            (short)0, acc[j], false, false);
        }
    }
#pragma unroll
    for (int j = 0; j < 4; ++j) {
        const int ncol = nBase + j * 16 + mr;
#pragma unroll
        for (int r = 0; r < 8; ++r) {
            int m = mBase + r + (half_ ? 8 : 0);   // C/D layout: VGPR r -> M=r / M=8+r
            float v = acc[j][r];
            if (mode & 1) {
                v += bias[ncol];
                v = (v > 20.f) ? v : __logf(1.f + __expf(v));   // softplus
            }
            C[(size_t)m * N + ncol] = v;
            if (mode & 2) C16[(size_t)m * N + ncol] = (_Float16)v;
        }
    }
}

// ---------------------------------------------------------------------------
// Depthwise causal conv (K=4) + SiLU in scan order (dir=1 time-reversed).
// ---------------------------------------------------------------------------
__global__ void vm_conv_silu(const float* __restrict__ xz,
                             const float* __restrict__ cw,
                             const float* __restrict__ cb,
                             float* __restrict__ uc,
                             _Float16* __restrict__ uc16, int dir) {
    int idx = blockIdx.x * blockDim.x + threadIdx.x;
    if (idx >= BL * DI) return;
    int d = idx % DI;
    int l = (idx / DI) % LL;
    int b = idx / (DI * LL);
    float s = cb[d];
#pragma unroll
    for (int t = 0; t < KK; ++t) {
        int ls = l - (KK - 1) + t;
        if (ls >= 0) {
            int lo = dir ? (LL - 1 - ls) : ls;
            s += xz[((size_t)(b * LL + lo)) * (2 * DI) + d] * cw[d * KK + t];
        }
    }
    float o = s / (1.f + __expf(-s));   // SiLU
    uc[idx] = o;
    uc16[idx] = (_Float16)o;
}

// ---------------------------------------------------------------------------
// Selective scan: thread per (b,d); h[16],A[16] in regs, B_t/C_t via LDS.
// Fuses +u*Dp, *silu(z), fwd/bwd merge; dir=1 also emits f16 ysum.
// ---------------------------------------------------------------------------
__global__ __launch_bounds__(256) void vm_scan(const float* __restrict__ uc,
                                               const float* __restrict__ dt,
                                               const float* __restrict__ xdbl,
                                               const float* __restrict__ Alog,
                                               const float* __restrict__ Dp,
                                               const float* __restrict__ xz,
                                               float* __restrict__ ysum,
                                               _Float16* __restrict__ ysum16, int dir) {
    __shared__ float Bs[SS], Cs[SS];
    int b = blockIdx.y;
    int d = blockIdx.x * 256 + threadIdx.x;
    float a[SS], h[SS];
#pragma unroll
    for (int s = 0; s < SS; ++s) { a[s] = -__expf(Alog[(size_t)d * SS + s]); h[s] = 0.f; }
    float dparam = Dp[d];
    for (int l = 0; l < LL; ++l) {
        size_t row = (size_t)b * LL + l;            // scan order
        if (threadIdx.x < 2 * SS) {
            float vv = xdbl[row * XDP + DTR + threadIdx.x];
            if (threadIdx.x < SS) Bs[threadIdx.x] = vv;
            else                  Cs[threadIdx.x - SS] = vv;
        }
        __syncthreads();
        float dtv = dt[row * DI + d];
        float uv  = uc[row * DI + d];
        float du  = dtv * uv;
        float y = 0.f;
#pragma unroll
        for (int s = 0; s < SS; ++s) {
            h[s] = __expf(dtv * a[s]) * h[s] + du * Bs[s];
            y += h[s] * Cs[s];
        }
        __syncthreads();
        int lo = dir ? (LL - 1 - l) : l;            // back to original order
        size_t orow = (size_t)b * LL + lo;
        float z = xz[orow * (2 * DI) + DI + d];
        float out = (y + uv * dparam) * (z / (1.f + __expf(-z)));
        if (dir) {
            float tot = ysum[orow * DI + d] + out;
            ysum[orow * DI + d] = tot;
            ysum16[orow * DI + d] = (_Float16)tot;
        } else {
            ysum[orow * DI + d] = out;
        }
    }
}

// ---------------------------------------------------------------------------
// Token mean over L, then classifier head.
// ---------------------------------------------------------------------------
__global__ void vm_mean(const float* __restrict__ feats, float* __restrict__ mb) {
    int idx = blockIdx.x * blockDim.x + threadIdx.x;
    if (idx >= BB * DD) return;
    int d = idx % DD, b = idx / DD;
    float s = 0.f;
    for (int l = 0; l < LL; ++l) s += feats[((size_t)b * LL + l) * DD + d];
    mb[idx] = s * (1.f / LL);
}

__global__ void vm_head(const float* __restrict__ mb,
                        const float* __restrict__ hw,
                        const float* __restrict__ hb,
                        float* __restrict__ out) {
    int idx = blockIdx.x * blockDim.x + threadIdx.x;
    if (idx >= BB * NCLS) return;
    int n = idx % NCLS, b = idx / NCLS;
    const float* mr = mb + (size_t)b * DD;
    const float* wr = hw + (size_t)n * DD;
    float s = hb[n];
    for (int d = 0; d < DD; ++d) s += mr[d] * wr[d];
    out[idx] = s;
}

// ---------------------------------------------------------------------------
extern "C" void kernel_launch(void* const* d_in, const int* in_sizes, int n_in,
                              void* d_out, int out_size, void* d_ws, size_t ws_size,
                              hipStream_t stream) {
    const float* x         = (const float*)d_in[0];
    const float* patch_w   = (const float*)d_in[1];
    const float* patch_b   = (const float*)d_in[2];
    const float* pos_embed = (const float*)d_in[3];
    const float* norm_w    = (const float*)d_in[4];
    const float* norm_b    = (const float*)d_in[5];
    const float* in_proj_w = (const float*)d_in[6];
    const float* conv_w    = (const float*)d_in[7];
    const float* conv_b    = (const float*)d_in[8];
    const float* x_proj_w  = (const float*)d_in[9];
    const float* dt_proj_w = (const float*)d_in[10];
    const float* dt_proj_b = (const float*)d_in[11];
    const float* A_log     = (const float*)d_in[12];
    const float* D_param   = (const float*)d_in[13];
    const float* out_proj_w= (const float*)d_in[14];
    const float* norm_f_w  = (const float*)d_in[15];
    const float* norm_f_b  = (const float*)d_in[16];
    const float* head_w    = (const float*)d_in[17];
    const float* head_b    = (const float*)d_in[18];
    float* out = (float*)d_out;

    char* ws = (char*)d_ws;
    size_t off = 0;
    auto allocB = [&](size_t bytes) {
        void* p = (void*)(ws + off);
        off += (bytes + 255) & ~(size_t)255;
        return p;
    };
    float* hidden   = (float*)allocB((size_t)BL * DD * 4);
    float* residual = (float*)allocB((size_t)BL * DD * 4);
    float* xn       = (float*)allocB((size_t)BL * DD * 4);
    float* xz       = (float*)allocB((size_t)BL * 2 * DI * 4);
    float* ucb      = (float*)allocB((size_t)BL * DI * 4);
    float* xdbl     = (float*)allocB((size_t)BL * XDP * 4);
    float* dtb      = (float*)allocB((size_t)BL * DI * 4);
    float* ysum     = (float*)allocB((size_t)BL * DI * 4);
    float* meanb    = (float*)allocB((size_t)BB * DD * 4);
    _Float16* xn16    = (_Float16*)allocB((size_t)BL * DD * 2);
    _Float16* uc16    = (_Float16*)allocB((size_t)BL * DI * 2);
    _Float16* xdbl16  = (_Float16*)allocB((size_t)BL * XDP * 2);
    _Float16* ysum16  = (_Float16*)allocB((size_t)BL * DI * 2);
    _Float16* inw16   = (_Float16*)allocB((size_t)2 * DI * DD * 2);
    _Float16* outw16  = (_Float16*)allocB((size_t)DD * DI * 2);
    _Float16* xpw16   = (_Float16*)allocB((size_t)2 * XDP * DI * 2);   // per-dir padded
    _Float16* dpw16   = (_Float16*)allocB((size_t)2 * DI * DTKP * 2);  // per-dir padded

    vm_patch_embed<<<(BL * DD + 255) / 256, 256, 0, stream>>>(x, patch_w, patch_b, pos_embed, hidden);
    vm_zero<<<(BL * DD + 255) / 256, 256, 0, stream>>>(residual, BL * DD);

    const int TM = BL / 16;                       // 49 exact M-tiles
    auto gy = [](int N) { return (N + 255) / 256; };  // 4 waves x 64 cols per block

    for (int layer = 0; layer < DEPTH; ++layer) {
        // --- per-layer weight prep (f16 + padding) ---
        vm_cvt_f16<<<(2 * DI * DD + 255) / 256, 256, 0, stream>>>(
            in_proj_w + (size_t)layer * 2 * DI * DD, inw16, 2 * DI * DD);
        vm_cvt_f16<<<(DD * DI + 255) / 256, 256, 0, stream>>>(
            out_proj_w + (size_t)layer * DD * DI, outw16, DD * DI);
        for (int dir = 0; dir < 2; ++dir) {
            size_t ld = (size_t)layer * 2 + dir;
            vm_pad_xproj<<<(XDP * DI + 255) / 256, 256, 0, stream>>>(
                x_proj_w + ld * (DTR + 2 * SS) * DI, xpw16 + (size_t)dir * XDP * DI);
            vm_pad_dtw<<<(DI * DTKP + 255) / 256, 256, 0, stream>>>(
                dt_proj_w + ld * DI * DTR, dpw16 + (size_t)dir * DI * DTKP);
        }

        vm_add_ln<<<BL, 128, 0, stream>>>(residual, hidden, xn, xn16,
                                          norm_w + (size_t)layer * DD,
                                          norm_b + (size_t)layer * DD, 1);
        // in_proj: [784,384] x [1536,384]^T -> xz
        vm_wmma_gemm<<<dim3(TM, gy(2 * DI)), 128, 0, stream>>>(
            xn16, inw16, nullptr, xz, nullptr, 2 * DI, DD, DD, 0);

        for (int dir = 0; dir < 2; ++dir) {
            size_t ld = (size_t)layer * 2 + dir;
            const float* cw  = conv_w    + ld * DI * KK;
            const float* cb  = conv_b    + ld * DI;
            const float* dpb = dt_proj_b + ld * DI;
            const float* Al  = A_log     + ld * DI * SS;
            const float* Dp  = D_param   + ld * DI;

            vm_conv_silu<<<(BL * DI + 255) / 256, 256, 0, stream>>>(xz, cw, cb, ucb, uc16, dir);
            // x_proj: [784,768] x [64,768]^T -> xdbl (f32 + f16 copy)
            vm_wmma_gemm<<<dim3(TM, gy(XDP)), 128, 0, stream>>>(
                uc16, xpw16 + (size_t)dir * XDP * DI, nullptr, xdbl, xdbl16,
                XDP, DI, DI, 2);
            // dt_proj: [784,32](lda=64) x [768,32]^T + bias, softplus -> dtb
            vm_wmma_gemm<<<dim3(TM, gy(DI)), 128, 0, stream>>>(
                xdbl16, dpw16 + (size_t)dir * DI * DTKP, dpb, dtb, nullptr,
                DI, DTKP, XDP, 1);

            vm_scan<<<dim3(DI / 256, BB), 256, 0, stream>>>(
                ucb, dtb, xdbl, Al, Dp, xz, ysum, ysum16, dir);
        }
        // out_proj: [784,768] x [384,768]^T -> hidden
        vm_wmma_gemm<<<dim3(TM, gy(DD)), 128, 0, stream>>>(
            ysum16, outw16, nullptr, hidden, nullptr, DD, DI, DI, 0);
    }

    vm_add_ln<<<BL, 128, 0, stream>>>(residual, hidden, xn, xn16, norm_f_w, norm_f_b, 0);
    vm_mean<<<(BB * DD + 255) / 256, 256, 0, stream>>>(xn, meanb);
    vm_head<<<(BB * NCLS + 255) / 256, 256, 0, stream>>>(meanb, head_w, head_b, out);
}
